// AV_Temporal_Attn_57509612093495
// MI455X (gfx1250) — compile-verified, matrix operand order checked by reading, not run on hardware
//
#include <hip/hip_runtime.h>
#include <hip/hip_bf16.h>
#include <stdint.h>

typedef __bf16 bf16_t;
typedef __attribute__((ext_vector_type(16))) __bf16 v16bf;
typedef __attribute__((ext_vector_type(8)))  __bf16 v8bf;
typedef __attribute__((ext_vector_type(4)))  __bf16 v4bf;
typedef __attribute__((ext_vector_type(8)))  float  v8f;

constexpr int kDIM   = 1024;
constexpr int kINNER = 2048;
constexpr int kB     = 8;
constexpr int kL     = 2048;
constexpr int kM     = kB * kL;     // 16384 rows
constexpr float kEPS = 1e-5f;

// ---------------------------------------------------------------------------
// CDNA5 async global->LDS copy (ASYNCcnt-tracked), per-lane 16B.
// ---------------------------------------------------------------------------
__device__ __forceinline__ void async_b128(uint32_t lds, const void* g) {
    unsigned long long ga = (unsigned long long)(uintptr_t)g;
    asm volatile("global_load_async_to_lds_b128 %0, %1, off"
                 :: "v"(lds), "v"(ga) : "memory");
}
__device__ __forceinline__ void wait_async_le8() {
    asm volatile("s_wait_asynccnt 8" ::: "memory");
}
__device__ __forceinline__ void wait_async_0() {
    asm volatile("s_wait_asynccnt 0" ::: "memory");
}

// ---------------------------------------------------------------------------
// Kernel 1: RMS-norm one row of [*, 1024] and cast to bf16.
// ---------------------------------------------------------------------------
__global__ __launch_bounds__(256) void k_rmsnorm_cast(
    const float* __restrict__ x, const float* __restrict__ w,
    bf16_t* __restrict__ y)
{
    const int row = blockIdx.x;
    const int t   = threadIdx.x;
    const float* xr = x + (size_t)row * kDIM;

    float4 xv = ((const float4*)xr)[t];
    float ss = xv.x * xv.x + xv.y * xv.y + xv.z * xv.z + xv.w * xv.w;

    for (int off = 16; off > 0; off >>= 1)
        ss += __shfl_down(ss, off, 32);

    __shared__ float wsum[8];
    const int wave = t >> 5, lane = t & 31;
    if (lane == 0) wsum[wave] = ss;
    __syncthreads();
    if (t == 0) {
        float s = 0.f;
        for (int i = 0; i < 8; ++i) s += wsum[i];
        wsum[0] = rsqrtf(s / (float)kDIM + kEPS);
    }
    __syncthreads();
    const float scale = wsum[0];

    const float4 wv = ((const float4*)w)[t];
    v4bf o;
    o[0] = (bf16_t)(xv.x * scale * wv.x);
    o[1] = (bf16_t)(xv.y * scale * wv.y);
    o[2] = (bf16_t)(xv.z * scale * wv.z);
    o[3] = (bf16_t)(xv.w * scale * wv.w);
    *(v4bf*)(y + (size_t)row * kDIM + t * 4) = o;
}

// ---------------------------------------------------------------------------
// Kernel 2: vectorized f32 -> bf16 cast (weights). n divisible by 1024.
// ---------------------------------------------------------------------------
__global__ __launch_bounds__(256) void k_cast_bf16(
    const float* __restrict__ x, bf16_t* __restrict__ y, int n)
{
    const int i = (blockIdx.x * blockDim.x + threadIdx.x) * 4;
    if (i < n) {
        const float4 v = *(const float4*)(x + i);
        v4bf o;
        o[0] = (bf16_t)v.x; o[1] = (bf16_t)v.y;
        o[2] = (bf16_t)v.z; o[3] = (bf16_t)v.w;
        *(v4bf*)(y + i) = o;
    }
}

// ---------------------------------------------------------------------------
// WMMA GEMM core: C[M,N] = A[M,K] * W[N,K]^T  (bf16 in, f32 accum)
// 128x128 block tile, BK=64 (two 16x16x32 k-subtiles per buffer step).
// 256 threads = 8 waves (4M x 2N); 8 acc tiles/wave; 32 WMMAs per loop body.
// Double-buffered LDS fed by global_load_async_to_lds_b128 (ASYNCcnt);
// buffer index is compile-time constant (loop unrolled x2) -> no selects.
// Epilogue staged through LDS for coalesced 16B global stores.
// MODE 0: bf16 C.  MODE 1: f32 residual + concat([a,v],axis=1) row remap.
// ---------------------------------------------------------------------------
template <int MODE>
__global__ __launch_bounds__(256) void k_gemm_wmma(
    const bf16_t* __restrict__ A, int K,
    const bf16_t* __restrict__ Wn,        // [N, K] row-major
    bf16_t* __restrict__ Cb, int ldc,     // MODE 0
    const float* __restrict__ Res,        // MODE 1
    float* __restrict__ Out,              // MODE 1: [kB, 2*kL, kDIM]
    int mod)
{
    constexpr int BK  = 64;
    constexpr int AST = 72;               // padded LDS row stride (144B rows)
    struct Tiles { bf16_t a[2][128 * AST]; bf16_t b[2][128 * AST]; };
    __shared__ union SMem {
        Tiles  t;
        bf16_t cb[128 * 136];             // MODE 0 output staging
        float  cf[128 * 132];             // MODE 1 output staging
    } sm;

    const int t     = threadIdx.x;
    const int wave  = t >> 5;
    const int lane  = t & 31;
    const int waveM = wave & 3;
    const int waveN = wave >> 2;
    const int bm    = blockIdx.y * 128;
    const int bn    = blockIdx.x * 128;

    v8f acc[2][4];
    for (int i = 0; i < 2; ++i)
        for (int j = 0; j < 4; ++j)
            for (int r = 0; r < 8; ++r) acc[i][j][r] = 0.f;

    // copy plan: 128x64 bf16 tile = 1024 x 16B chunks; 4 chunks/thread
    const bf16_t* gAp[4];
    const bf16_t* gBp[4];
    uint32_t loff[4];
    for (int p = 0; p < 4; ++p) {
        const int idx = t + p * 256;
        const int row = idx >> 3;         // 0..127
        const int col = (idx & 7) * 8;    // 0,8,...,56 (elems)
        gAp[p]  = A  + (size_t)(bm + row) * K + col;
        gBp[p]  = Wn + (size_t)(bn + row) * K + col;
        loff[p] = (uint32_t)((row * AST + col) * sizeof(bf16_t));
    }
    const uint32_t baseA0 = (uint32_t)(uintptr_t)&sm.t.a[0][0];
    const uint32_t baseA1 = (uint32_t)(uintptr_t)&sm.t.a[1][0];
    const uint32_t baseB0 = (uint32_t)(uintptr_t)&sm.t.b[0][0];
    const uint32_t baseB1 = (uint32_t)(uintptr_t)&sm.t.b[1][0];

    auto issue = [&](uint32_t bA, uint32_t bB, int kt) {
        #pragma unroll
        for (int p = 0; p < 4; ++p) async_b128(bA + loff[p], gAp[p] + kt);
        #pragma unroll
        for (int p = 0; p < 4; ++p) async_b128(bB + loff[p], gBp[p] + kt);
    };

    const int half = lane >> 4;           // K-octet selector
    const int r16  = lane & 15;           // fragment row (M or N)

    auto compute = [&](const bf16_t* __restrict__ aT, const bf16_t* __restrict__ bT) {
        #pragma unroll
        for (int ks = 0; ks < 2; ++ks) {
            const int kof = ks * 32;
            v16bf af[2];
            #pragma unroll
            for (int i = 0; i < 2; ++i) {
                const bf16_t* p = aT + (waveM * 32 + i * 16 + r16) * AST + kof;
                v8bf lo = *(const v8bf*)(p + half * 8);
                v8bf hi = *(const v8bf*)(p + 16 + half * 8);
                af[i] = __builtin_shufflevector(lo, hi,
                        0,1,2,3,4,5,6,7,8,9,10,11,12,13,14,15);
            }
            v16bf bfr[4];
            #pragma unroll
            for (int j = 0; j < 4; ++j) {
                const bf16_t* p = bT + (waveN * 64 + j * 16 + r16) * AST + kof;
                v8bf lo = *(const v8bf*)(p + half * 8);
                v8bf hi = *(const v8bf*)(p + 16 + half * 8);
                bfr[j] = __builtin_shufflevector(lo, hi,
                        0,1,2,3,4,5,6,7,8,9,10,11,12,13,14,15);
            }
            #pragma unroll
            for (int i = 0; i < 2; ++i)
                #pragma unroll
                for (int j = 0; j < 4; ++j)
                    acc[i][j] = __builtin_amdgcn_wmma_f32_16x16x32_bf16(
                        false, af[i], false, bfr[j], (short)0, acc[i][j],
                        false, false);
        }
    };

    issue(baseA0, baseB0, 0);
    for (int kt = 0; kt < K; kt += 2 * BK) {
        issue(baseA1, baseB1, kt + BK);     // overlap with buf0 compute
        wait_async_le8();                   // buf0's 8 copies done (in-order)
        __syncthreads();
        compute(&sm.t.a[0][0], &sm.t.b[0][0]);
        __syncthreads();                    // buf0 reads done before refill

        if (kt + 2 * BK < K) {
            issue(baseA0, baseB0, kt + 2 * BK);
            wait_async_le8();               // buf1's copies done
        } else {
            wait_async_0();
        }
        __syncthreads();
        compute(&sm.t.a[1][0], &sm.t.b[1][0]);
        __syncthreads();                    // buf1 reads done before refill
    }

    // ------------------ epilogue: stage tile in LDS, coalesced stores -----
    const int n0 = lane & 15;
    if (MODE == 0) {
        for (int i = 0; i < 2; ++i)
            for (int j = 0; j < 4; ++j)
                for (int r = 0; r < 8; ++r) {
                    const int row = waveM * 32 + i * 16 + half * 8 + r;
                    const int col = waveN * 64 + j * 16 + n0;
                    sm.cb[row * 136 + col] = (bf16_t)acc[i][j][r];
                }
        __syncthreads();
        for (int it = 0; it < 8; ++it) {
            const int idx = t + it * 256;         // 2048 16B-chunks
            const int row = idx >> 4;
            const int ch  = idx & 15;
            *(uint4*)(Cb + (size_t)(bm + row) * ldc + bn + ch * 8) =
                *(const uint4*)&sm.cb[row * 136 + ch * 8];
        }
    } else {
        for (int i = 0; i < 2; ++i)
            for (int j = 0; j < 4; ++j)
                for (int r = 0; r < 8; ++r) {
                    const int row = waveM * 32 + i * 16 + half * 8 + r;
                    const int col = waveN * 64 + j * 16 + n0;
                    sm.cf[row * 132 + col] = acc[i][j][r];
                }
        __syncthreads();
        for (int it = 0; it < 16; ++it) {
            const int idx = t + it * 256;         // 4096 16B-chunks
            const int row = idx >> 5;
            const int ch  = idx & 31;
            const int grow = bm + row;
            const int bb = grow >> 11;            // / kL
            const int ll = grow & (kL - 1);
            const size_t orow = (size_t)bb * (2 * kL) + (size_t)mod * kL + ll;
            float4 r4 = *(const float4*)(Res + (size_t)grow * kDIM + bn + ch * 4);
            const float4 c4 = *(const float4*)&sm.cf[row * 132 + ch * 4];
            r4.x += c4.x; r4.y += c4.y; r4.z += c4.z; r4.w += c4.w;
            *(float4*)(Out + orow * kDIM + bn + ch * 4) = r4;
        }
    }
}

// ---------------------------------------------------------------------------
// Kernel 4: causal dwconv (K=4) + cross-modal SiLU gating, 8 channels/thread.
// ---------------------------------------------------------------------------
__device__ __forceinline__ float silu_f(float x) {
    return x / (1.0f + __expf(-x));
}

__global__ __launch_bounds__(256) void k_conv_gate(
    const bf16_t* __restrict__ projA, const bf16_t* __restrict__ projV,
    const float* __restrict__ wa, const float* __restrict__ ba,
    const float* __restrict__ wv, const float* __restrict__ bv,
    bf16_t* __restrict__ gatedA, bf16_t* __restrict__ gatedV)
{
    const size_t idx = (size_t)blockIdx.x * blockDim.x + threadIdx.x;
    const int e8 = (int)(idx & (kINNER / 8 - 1)) * 8;   // channel base
    const int m  = (int)(idx >> 8);                     // kINNER/8 = 256
    const int l  = m & (kL - 1);
    const size_t rowStride = 2 * kINNER;

    v8bf xa[4], xv[4];
    #pragma unroll
    for (int j = 0; j < 4; ++j) {
        if (l - 3 + j >= 0) {
            const size_t off = (size_t)(m - 3 + j) * rowStride + e8;
            xa[j] = *(const v8bf*)(projA + off);
            xv[j] = *(const v8bf*)(projV + off);
        } else {
            #pragma unroll
            for (int c = 0; c < 8; ++c) { xa[j][c] = (bf16_t)0.f; xv[j][c] = (bf16_t)0.f; }
        }
    }
    const v8bf gAraw = *(const v8bf*)(projA + (size_t)m * rowStride + kINNER + e8);
    const v8bf gVraw = *(const v8bf*)(projV + (size_t)m * rowStride + kINNER + e8);

    v8bf outA, outV;
    #pragma unroll
    for (int c = 0; c < 8; ++c) {
        const float4 wA4 = *(const float4*)(wa + (size_t)(e8 + c) * 4);
        const float4 wV4 = *(const float4*)(wv + (size_t)(e8 + c) * 4);
        float sA = ba[e8 + c];
        float sV = bv[e8 + c];
        sA += wA4.x * (float)xa[0][c] + wA4.y * (float)xa[1][c]
            + wA4.z * (float)xa[2][c] + wA4.w * (float)xa[3][c];
        sV += wV4.x * (float)xv[0][c] + wV4.y * (float)xv[1][c]
            + wV4.z * (float)xv[2][c] + wV4.w * (float)xv[3][c];
        outA[c] = (bf16_t)(sA * silu_f((float)gVraw[c]));   // a_x * silu(v_w)
        outV[c] = (bf16_t)(sV * silu_f((float)gAraw[c]));   // v_x * silu(a_w)
    }
    *(v8bf*)(gatedA + (size_t)m * kINNER + e8) = outA;
    *(v8bf*)(gatedV + (size_t)m * kINNER + e8) = outV;
}

// ---------------------------------------------------------------------------
// Launch
// ---------------------------------------------------------------------------
extern "C" void kernel_launch(void* const* d_in, const int* in_sizes, int n_in,
                              void* d_out, int out_size, void* d_ws, size_t ws_size,
                              hipStream_t stream) {
    const float* audio    = (const float*)d_in[0];
    const float* visual   = (const float*)d_in[1];
    const float* a_norm_w = (const float*)d_in[2];
    const float* v_norm_w = (const float*)d_in[3];
    const float* a_in_w   = (const float*)d_in[4];
    const float* v_in_w   = (const float*)d_in[5];
    const float* a_conv_w = (const float*)d_in[6];
    const float* a_conv_b = (const float*)d_in[7];
    const float* v_conv_w = (const float*)d_in[8];
    const float* v_conv_b = (const float*)d_in[9];
    const float* a_out_w  = (const float*)d_in[10];
    const float* v_out_w  = (const float*)d_in[11];
    float* out = (float*)d_out;

    char* ws = (char*)d_ws;
    size_t off = 0;
    auto carve = [&](size_t elems) -> bf16_t* {
        bf16_t* p = (bf16_t*)(ws + off);
        off += ((elems * sizeof(bf16_t)) + 255) & ~(size_t)255;
        return p;
    };
    bf16_t* nA     = carve((size_t)kM * kDIM);
    bf16_t* nV     = carve((size_t)kM * kDIM);
    bf16_t* WinA   = carve((size_t)2 * kINNER * kDIM);
    bf16_t* WinV   = carve((size_t)2 * kINNER * kDIM);
    bf16_t* WoutA  = carve((size_t)kDIM * kINNER);
    bf16_t* WoutV  = carve((size_t)kDIM * kINNER);
    bf16_t* projA  = carve((size_t)kM * 2 * kINNER);
    bf16_t* projV  = carve((size_t)kM * 2 * kINNER);
    bf16_t* gatedA = carve((size_t)kM * kINNER);
    bf16_t* gatedV = carve((size_t)kM * kINNER);

    // 1) RMS-norm + cast
    k_rmsnorm_cast<<<kM, 256, 0, stream>>>(audio,  a_norm_w, nA);
    k_rmsnorm_cast<<<kM, 256, 0, stream>>>(visual, v_norm_w, nV);

    // 2) weight casts
    {
        const int nIn  = 2 * kINNER * kDIM;
        const int nOut = kDIM * kINNER;
        k_cast_bf16<<<nIn  / 1024, 256, 0, stream>>>(a_in_w,  WinA,  nIn);
        k_cast_bf16<<<nIn  / 1024, 256, 0, stream>>>(v_in_w,  WinV,  nIn);
        k_cast_bf16<<<nOut / 1024, 256, 0, stream>>>(a_out_w, WoutA, nOut);
        k_cast_bf16<<<nOut / 1024, 256, 0, stream>>>(v_out_w, WoutV, nOut);
    }

    // 3) input projection GEMMs
    {
        dim3 grid(2 * kINNER / 128, kM / 128);   // (32, 128)
        k_gemm_wmma<0><<<grid, 256, 0, stream>>>(nA, kDIM, WinA, projA, 2 * kINNER,
                                                 nullptr, nullptr, 0);
        k_gemm_wmma<0><<<grid, 256, 0, stream>>>(nV, kDIM, WinV, projV, 2 * kINNER,
                                                 nullptr, nullptr, 0);
    }

    // 4) causal dwconv + cross SiLU gating
    {
        const size_t total = (size_t)kM * kINNER / 8;  // 4,194,304 threads
        k_conv_gate<<<(unsigned)(total / 256), 256, 0, stream>>>(
            projA, projV, a_conv_w, a_conv_b, v_conv_w, v_conv_b, gatedA, gatedV);
    }

    // 5) output GEMMs + residual, concat([a,v], axis=1) layout
    {
        dim3 grid(kDIM / 128, kM / 128);          // (8, 128)
        k_gemm_wmma<1><<<grid, 256, 0, stream>>>(gatedA, kINNER, WoutA, nullptr, 0,
                                                 audio, out, 0);
        k_gemm_wmma<1><<<grid, 256, 0, stream>>>(gatedV, kINNER, WoutV, nullptr, 0,
                                                 visual, out, 1);
    }
}
